// CFConv_22007412425372
// MI455X (gfx1250) — matrix operand order repaired and checked by reading, not compiled, not run on hardware
//
#include <hip/hip_runtime.h>

// CDNA5 / gfx1250 wave32 WMMA types
typedef __attribute__((ext_vector_type(2)))  float    v2f;
typedef __attribute__((ext_vector_type(8)))  float    v8f;
typedef __attribute__((ext_vector_type(8)))  _Float16 v8h;
typedef __attribute__((ext_vector_type(16))) _Float16 v16h;

__device__ __forceinline__ float ssp_f(float x) {
  // shifted softplus, numerically stable
  return fmaxf(x, 0.0f) + log1pf(expf(-fabsf(x))) - 0.69314718055994531f;
}

__device__ __forceinline__ v8f vzero8() {
  v8f z = {0.f, 0.f, 0.f, 0.f, 0.f, 0.f, 0.f, 0.f};
  return z;
}

// ---------------------------------------------------------------------------
// FP32 GEMM: O[64x128] = X[64x128] @ W[128x128] (+bias) (+ssp)
// one block = 64 output rows; 8 waves, each wave: 1 row-tile x 4 col-tiles.
// Uses v_wmma_f32_16x16x4_f32 (full precision path).
// Loads are batched per k-step so the 4-WMMA chain runs with one DS wait.
// ---------------------------------------------------------------------------
__global__ void __launch_bounds__(256)
gemm64_f32_kernel(const float* __restrict__ Xg, const float* __restrict__ Wg,
                  const float* __restrict__ bias, float* __restrict__ Og,
                  int act) {
  extern __shared__ char smemraw[];
  float* sX = (float*)smemraw;                      // 64 x 132 (pad: conflict-free A reads)
  float* sW = (float*)(smemraw + 64 * 132 * 4);     // 128 x 128
  const int tid = threadIdx.x;
  const size_t blk = (size_t)blockIdx.x * 64 * 128;

  for (int i = tid; i < 64 * 32; i += 256) {        // 2048 float4
    int r = i >> 5, c4 = (i & 31) << 2;
    float4 v = ((const float4*)(Xg + blk))[i];
    *(float4*)(sX + r * 132 + c4) = v;
  }
  for (int i = tid; i < 128 * 32; i += 256)         // 4096 float4
    ((float4*)sW)[i] = ((const float4*)Wg)[i];
  __syncthreads();

  const int wave = tid >> 5, lane = tid & 31;
  const int mt = wave >> 1, ntb = (wave & 1) * 4;
  const int row  = mt * 16 + (lane & 15);
  const int ksel = (lane >> 4) << 1;                // f32 A: lanes 16-31 hold K+2,K+3
  const int cl   = lane & 15;

  v8f acc[4] = {vzero8(), vzero8(), vzero8(), vzero8()};
  for (int k0 = 0; k0 < 128; k0 += 4) {
    // batch all fragment loads, then the WMMA chain
    v2f a = *(const v2f*)(sX + row * 132 + k0 + ksel);   // 8B ds load
    v2f b[4];
#pragma unroll
    for (int i = 0; i < 4; ++i) {
      int n = (ntb + i) * 16 + cl;
      b[i].x = sW[(k0 + ksel) * 128 + n];
      b[i].y = sW[(k0 + ksel + 1) * 128 + n];
    }
#pragma unroll
    for (int i = 0; i < 4; ++i)
      acc[i] = __builtin_amdgcn_wmma_f32_16x16x4_f32(
          false, a, false, b[i], (short)0, acc[i], false, false);
  }

  const int rsel = (lane >> 4) << 3;                // D: lanes 16-31 hold rows M+8
#pragma unroll
  for (int i = 0; i < 4; ++i) {
    int col = (ntb + i) * 16 + cl;
    float bv = bias ? bias[col] : 0.0f;
#pragma unroll
    for (int r = 0; r < 8; ++r) {
      int ro = mt * 16 + rsel + r;
      float v = acc[i][r] + bv;
      if (act) v = ssp_f(v);
      Og[blk + (size_t)ro * 128 + col] = v;
    }
  }
}

// ---------------------------------------------------------------------------
// Fused CFConv core: one block per (b,a).
//  stage1: H[64x128]  = ssp(fij[64x50 padK->64] @ Wf1 + b1)   (f16 WMMA)
//  stage2: Wf[64x128] = H @ Wf2 + b2                          (f16 WMMA)
//  stage3: y[f] = sum_n mask[n] * xW[b, nb[n], f] * Wf[n,f]
// A fragments: two 16B ds_load_b128 + shufflevector (no per-element VALU).
// B matrices pre-packed in fragment order -> one 32B load per lane per tile.
// ---------------------------------------------------------------------------
__global__ void __launch_bounds__(256)
cfconv_fused_kernel(const float* __restrict__ fij, const int* __restrict__ nbrs,
                    const float* __restrict__ pmask,
                    const float* __restrict__ Wf1, const float* __restrict__ bf1,
                    const float* __restrict__ Wf2, const float* __restrict__ bf2,
                    const float* __restrict__ xW, float* __restrict__ yBuf) {
  extern __shared__ char smemraw[];
  // scratch region [0, 32768): stage1 = sFijH + packed Wf1; stage2 = packed Wf2;
  //                            stage3 = 256-float partial sums
  _Float16* sFijH   = (_Float16*)smemraw;           // 64 x 72 halves (9216 B)
  _Float16* sBp1    = (_Float16*)(smemraw + 9216);  // packed Wf1: 64x128 (16384 B)
  _Float16* sBp2    = (_Float16*)smemraw;           // packed Wf2: 128x128 (32768 B)
  float*    partial = (float*)smemraw;              // 256 floats
  _Float16* sHH     = (_Float16*)(smemraw + 32768); // 64 x 136 halves (17408 B)
  float*    sWf     = (float*)(smemraw + 50176);    // 64 x 132 floats (33792 B)
  int*      sNb     = (int*)(smemraw + 83968);      // 64
  float*    sMask   = (float*)(smemraw + 84224);    // 64

  const int tid = threadIdx.x;
  const int bid = blockIdx.x;                       // = b*1024 + a

  if (tid < 64) {
    sNb[tid]   = nbrs[(size_t)bid * 64 + tid];
    sMask[tid] = pmask[(size_t)bid * 64 + tid];
  }
  // fij tile, K padded 50 -> 64 with zeros (stride 72: 16B-aligned, conflict-free)
  const float* fblk = fij + (size_t)bid * 64 * 50;
  for (int i = tid; i < 64 * 72; i += 256) {
    int n = i / 72, g = i - n * 72;
    sFijH[i] = (g < 50) ? (_Float16)fblk[n * 50 + g] : (_Float16)0.0f;
  }
  // Wf1 packed into B-fragment order, zero rows k>=50
  for (int p = tid; p < 64 * 128; p += 256) {
    int e = p & 15, ln = (p >> 4) & 31, nt = (p >> 9) & 7, kblk = p >> 12;
    int k = kblk * 32 + ((e >> 3) << 4) + ((ln >> 4) << 3) + (e & 7);
    int n = nt * 16 + (ln & 15);
    sBp1[p] = (k < 50) ? (_Float16)Wf1[k * 128 + n] : (_Float16)0.0f;
  }
  __syncthreads();

  const int wave = tid >> 5, lane = tid & 31;
  const int mt = wave >> 1, ntb = (wave & 1) * 4;
  const int row  = mt * 16 + (lane & 15);
  const int cl   = lane & 15;
  const int kbA  = (lane >> 4) * 8;                 // f16 A: hi lanes hold K+8..15 / K+24..31
  const int rsel = (lane >> 4) << 3;

  // ---- Stage 1: K = 64 (2 x v_wmma_f32_16x16x32_f16 k-steps per tile)
  {
    const v16h* bp = (const v16h*)sBp1;
    v8f acc[4] = {vzero8(), vzero8(), vzero8(), vzero8()};
#pragma unroll
    for (int k0 = 0; k0 < 64; k0 += 32) {
      const int kblk = k0 >> 5;
      v8h lo = *(const v8h*)(sFijH + row * 72 + k0 + kbA);
      v8h hi = *(const v8h*)(sFijH + row * 72 + k0 + 16 + kbA);
      v16h a = __builtin_shufflevector(lo, hi, 0, 1, 2, 3, 4, 5, 6, 7,
                                               8, 9, 10, 11, 12, 13, 14, 15);
      v16h b[4];
#pragma unroll
      for (int i = 0; i < 4; ++i)
        b[i] = bp[(kblk * 8 + (ntb + i)) * 32 + lane];
#pragma unroll
      for (int i = 0; i < 4; ++i)
        acc[i] = __builtin_amdgcn_wmma_f32_16x16x32_f16(
            false, a, false, b[i], (short)0, acc[i], false, false);
    }
#pragma unroll
    for (int i = 0; i < 4; ++i) {
      int col = (ntb + i) * 16 + cl;
      float bv = bf1[col];
#pragma unroll
      for (int r = 0; r < 8; ++r)
        sHH[(mt * 16 + rsel + r) * 136 + col] = (_Float16)ssp_f(acc[i][r] + bv);
    }
  }
  __syncthreads();

  // pack Wf2 (overwrites stage-1 scratch)
  for (int p = tid; p < 128 * 128; p += 256) {
    int e = p & 15, ln = (p >> 4) & 31, nt = (p >> 9) & 7, kblk = p >> 12;
    int k = kblk * 32 + ((e >> 3) << 4) + ((ln >> 4) << 3) + (e & 7);
    int n = nt * 16 + (ln & 15);
    sBp2[p] = (_Float16)Wf2[k * 128 + n];
  }
  __syncthreads();

  // ---- Stage 2: K = 128
  {
    const v16h* bp = (const v16h*)sBp2;
    v8f acc[4] = {vzero8(), vzero8(), vzero8(), vzero8()};
#pragma unroll
    for (int k0 = 0; k0 < 128; k0 += 32) {
      const int kblk = k0 >> 5;
      v8h lo = *(const v8h*)(sHH + row * 136 + k0 + kbA);
      v8h hi = *(const v8h*)(sHH + row * 136 + k0 + 16 + kbA);
      v16h a = __builtin_shufflevector(lo, hi, 0, 1, 2, 3, 4, 5, 6, 7,
                                               8, 9, 10, 11, 12, 13, 14, 15);
      v16h b[4];
#pragma unroll
      for (int i = 0; i < 4; ++i)
        b[i] = bp[(kblk * 8 + (ntb + i)) * 32 + lane];
#pragma unroll
      for (int i = 0; i < 4; ++i)
        acc[i] = __builtin_amdgcn_wmma_f32_16x16x32_f16(
            false, a, false, b[i], (short)0, acc[i], false, false);
    }
#pragma unroll
    for (int i = 0; i < 4; ++i) {
      int col = (ntb + i) * 16 + cl;
      float bv = bf2[col];
#pragma unroll
      for (int r = 0; r < 8; ++r)
        sWf[(mt * 16 + rsel + r) * 132 + col] = acc[i][r] + bv;
    }
  }
  __syncthreads();

  // ---- Stage 3: gather precomputed xW rows, gate by filter, sum over neighbors
  {
    const int f = tid & 127, half = tid >> 7;
    const float* xWb = xW + (size_t)(bid >> 10) * 1024 * 128;  // batch b
    float a3 = 0.0f;
#pragma unroll 4
    for (int i = 0; i < 32; ++i) {
      int n = half * 32 + i;
      a3 += sMask[n] * xWb[(size_t)sNb[n] * 128 + f] * sWf[n * 132 + f];
    }
    partial[tid] = a3;
  }
  __syncthreads();
  if (tid < 128)
    yBuf[(size_t)bid * 128 + tid] = partial[tid] + partial[tid + 128];
}

// ---------------------------------------------------------------------------
extern "C" void kernel_launch(void* const* d_in, const int* in_sizes, int n_in,
                              void* d_out, int out_size, void* d_ws, size_t ws_size,
                              hipStream_t stream) {
  (void)in_sizes; (void)n_in; (void)out_size; (void)ws_size;
  const float* x      = (const float*)d_in[0];
  const float* pmask  = (const float*)d_in[1];
  const int*   nbrs   = (const int*)d_in[2];
  const float* fij    = (const float*)d_in[3];
  const float* Wf1    = (const float*)d_in[4];
  const float* bf1    = (const float*)d_in[5];
  const float* Wf2    = (const float*)d_in[6];
  const float* bf2    = (const float*)d_in[7];
  const float* Win2f  = (const float*)d_in[8];
  const float* Wf2out = (const float*)d_in[9];
  const float* bf2out = (const float*)d_in[10];
  float* out = (float*)d_out;

  float* xWbuf = (float*)d_ws;                       // 8192*128 f32 = 4 MB
  float* yBuf  = xWbuf + (size_t)8192 * 128;         // 8192*128 f32 = 4 MB

  const size_t ldsGemm  = (64 * 132 + 128 * 128) * sizeof(float);  // 99328 B
  const size_t ldsFused = 84480;                                   // bytes

  // 1) xW = x @ W_in2f  (gather-after-dense trick; fp32 WMMA)
  gemm64_f32_kernel<<<128, 256, ldsGemm, stream>>>(x, Win2f, nullptr, xWbuf, 0);
  // 2) fused filter-net + gather + gated neighbor-sum (f16 WMMA, fp32 accum)
  cfconv_fused_kernel<<<8192, 256, ldsFused, stream>>>(
      fij, nbrs, pmask, Wf1, bf1, Wf2, bf2, xWbuf, yBuf);
  // 3) out = ssp(y @ W_f2out + b)  (fp32 WMMA)
  gemm64_f32_kernel<<<128, 256, ldsGemm, stream>>>(yBuf, Wf2out, bf2out, out, 1);
}
